// DownLayer_2370821947964
// MI455X (gfx1250) — matrix operand order repaired and checked by reading, not compiled, not run on hardware
//
#include <hip/hip_runtime.h>

typedef __attribute__((ext_vector_type(16))) _Float16 v16h;
typedef __attribute__((ext_vector_type(8)))  _Float16 v8h;
typedef __attribute__((ext_vector_type(8)))  float    v8f;

// ---------------- problem constants ----------------
constexpr int B_    = 32;
constexpr int N_    = 3136;
constexpr int CIN   = 128;
constexpr int COUT  = 320;
constexpr int H_    = 56;
constexpr int W_    = 56;
constexpr int HW_   = H_ * W_;     // 3136
constexpr int HO    = 28;
constexpr int WO    = 28;
constexpr int MO    = HO * WO;     // 784
constexpr int KC    = CIN * 9;     // 1152
constexpr int NGRID = 196;
constexpr int NADA  = N_ - NGRID;  // 2940
constexpr int SNUM  = 588;         // ceil(3136*0.25) - 196
constexpr int C1    = CIN + 1;     // 129

// normalized 1-D Gaussian, ksize=9 sigma=2 (separable; outer product == ref 2D kernel)
__constant__ float GW[9] = {0.0276305f, 0.0662822f, 0.1238314f, 0.1801736f,
                            0.2041637f, 0.1801736f, 0.1238314f, 0.0662822f,
                            0.0276305f};

#define CDIV(a, b) (((a) + (b) - 1) / (b))

// ---------------- 0: zero accumulator ----------------
__global__ void zero_kernel(float* p, size_t n) {
  size_t t = (size_t)blockIdx.x * blockDim.x + threadIdx.x;
  if (t < n) p[t] = 0.f;
}

// ---------------- 1: scatter tokens -> (B,HW,129) ----------------
__global__ void scatter_kernel(const float* __restrict__ x,
                               const float* __restrict__ pos,
                               float* __restrict__ accum) {
  size_t t = (size_t)blockIdx.x * blockDim.x + threadIdx.x;
  size_t total = (size_t)B_ * N_ * C1;
  if (t >= total) return;
  size_t bn = t / C1;
  int c = (int)(t % C1);
  int b = (int)(bn / N_);
  float px = fminf(fmaxf(pos[bn * 2 + 0], -1.f), 1.f);
  float py = fminf(fmaxf(pos[bn * 2 + 1], -1.f), 1.f);
  int lx = (int)rintf(0.5f * (px + 1.f) * W_ - 0.5f);  // RTE like jnp.round
  int ly = (int)rintf(0.5f * (py + 1.f) * H_ - 0.5f);
  lx = min(max(lx, 0), W_ - 1);
  ly = min(max(ly, 0), H_ - 1);
  size_t dst = ((size_t)b * HW_ + (size_t)ly * W_ + lx) * C1 + c;
  float v = (c < CIN) ? x[bn * CIN + c] : 1.0f;
  atomicAdd(accum + dst, v);
}

// ---------------- 2: normalize (accum -> norm) ----------------
__global__ void normalize_kernel(const float* __restrict__ accum,
                                 float* __restrict__ nrm) {
  size_t t = (size_t)blockIdx.x * blockDim.x + threadIdx.x;
  size_t total = (size_t)B_ * HW_ * C1;
  if (t >= total) return;
  size_t pix = t / C1;
  int c = (int)(t % C1);
  float m = accum[pix * C1 + CIN];
  float mb = (m > 0.f) ? 1.f : 0.f;
  if (c < CIN) {
    nrm[t] = accum[t] / (m + 1e-6f) * mb;
  } else {
    nrm[t] = mb;
  }
}

// ---------------- 3: horizontal 9-tap (nrm -> hbuf) ----------------
__global__ void hfilt_kernel(const float* __restrict__ in, float* __restrict__ out) {
  size_t t = (size_t)blockIdx.x * blockDim.x + threadIdx.x;
  size_t total = (size_t)B_ * HW_ * C1;
  if (t >= total) return;
  size_t pix = t / C1;
  int c = (int)(t % C1);
  int xw = (int)(pix % W_);
  float acc = 0.f;
#pragma unroll
  for (int d = -4; d <= 4; ++d) {
    int xx = xw + d;
    if (xx >= 0 && xx < W_) acc += GW[d + 4] * in[(pix + d) * C1 + c];
  }
  out[t] = acc;
}

// ---------------- 4: vertical 9-tap + hole fill -> f16 (B,HW,128) NHWC --------
__global__ void vfilt_combine_kernel(const float* __restrict__ hbuf,
                                     const float* __restrict__ nrm,
                                     _Float16* __restrict__ feat) {
  size_t t = (size_t)blockIdx.x * blockDim.x + threadIdx.x;
  size_t total = (size_t)B_ * HW_ * CIN;
  if (t >= total) return;
  size_t pix = t / CIN;
  int c = (int)(t % CIN);
  int yw = (int)((pix % HW_) / W_);
  float fi = 0.f, mi = 0.f;
#pragma unroll
  for (int d = -4; d <= 4; ++d) {
    int yy = yw + d;
    if (yy >= 0 && yy < H_) {
      size_t q = pix + (size_t)d * W_;
      fi += GW[d + 4] * hbuf[q * C1 + c];
      mi += GW[d + 4] * hbuf[q * C1 + CIN];
    }
  }
  fi = fi / fmaxf(mi, 1e-6f);
  fi = (mi > 0.f) ? fi : 0.f;
  float f0 = nrm[pix * C1 + c];
  float m0 = nrm[pix * C1 + CIN];
  feat[t] = (_Float16)(f0 + (1.f - m0) * fi);
}

// ---------------- 5: pack weights -> f16 [n][k], k=(kh*3+kw)*128+ci ----------------
__global__ void packw_kernel(const float* __restrict__ w, _Float16* __restrict__ pw) {
  int t = blockIdx.x * blockDim.x + threadIdx.x;
  int total = COUT * KC;
  if (t >= total) return;
  int n = t / KC;
  int k = t % KC;
  int khw = k >> 7;          // 0..8
  int ci = k & 127;
  pw[t] = (_Float16)w[((size_t)n * CIN + ci) * 9 + khw];
}

// ---------------- 6: implicit-im2col GEMM conv via WMMA ----------------
// feat:(B,56,56,128) NHWC f16   pw:(320,1152) f16   out:(B,784,320) f32
// One wave computes a 16x64 output tile: one A fragment shared by 4 B fragments.
__global__ __launch_bounds__(32) void conv_wmma_kernel(
    const _Float16* __restrict__ feat, const _Float16* __restrict__ pw,
    const float* __restrict__ bias, float* __restrict__ out) {
  const int lane = threadIdx.x;
  const int half = lane >> 4;
  const int l16 = lane & 15;
  const int b = blockIdx.y;
  const int tile_j = blockIdx.x % (COUT / 64);   // 0..4
  const int tile_m = blockIdx.x / (COUT / 64);   // 0..48
  const int m = tile_m * 16 + l16;               // A-row for this lane
  const int oh = m / WO, ow = m % WO;
  const int nb = tile_j * 64 + l16;              // base B-column for this lane

  v8f acc0 = {0.f, 0.f, 0.f, 0.f, 0.f, 0.f, 0.f, 0.f};
  v8f acc1 = acc0, acc2 = acc0, acc3 = acc0;

  for (int kk = 0; kk < KC; kk += 32) {
    v16h a;
    // A fragment: two 8-element K sub-blocks per lane (contiguous K: lane half 0-15/16-31)
#pragma unroll
    for (int blk = 0; blk < 2; ++blk) {
      int k0 = kk + (half ? 16 : 0) + blk * 8;
      int khw = k0 >> 7;     // constant across the 8 (32 | 128)
      int ci = k0 & 127;
      int kh = khw / 3, kw = khw % 3;
      int ih = 2 * oh + kh - 1;
      int iw = 2 * ow + kw - 1;
      v8h chunk = {(_Float16)0, (_Float16)0, (_Float16)0, (_Float16)0,
                   (_Float16)0, (_Float16)0, (_Float16)0, (_Float16)0};
      if (ih >= 0 && ih < H_ && iw >= 0 && iw < W_) {
        chunk = *(const v8h*)(feat + (((size_t)b * H_ + ih) * W_ + iw) * CIN + ci);
      }
#pragma unroll
      for (int e = 0; e < 8; ++e) a[blk * 8 + e] = chunk[e];
    }
    // B fragments: 16 consecutive K for each of 4 columns (lane half: K 0-15/16-31)
    int kb = kk + (half ? 16 : 0);
    v16h bf0 = *(const v16h*)(pw + (size_t)(nb +  0) * KC + kb);
    v16h bf1 = *(const v16h*)(pw + (size_t)(nb + 16) * KC + kb);
    v16h bf2 = *(const v16h*)(pw + (size_t)(nb + 32) * KC + kb);
    v16h bf3 = *(const v16h*)(pw + (size_t)(nb + 48) * KC + kb);

    acc0 = __builtin_amdgcn_wmma_f32_16x16x32_f16(false, a, false, bf0, (short)0, acc0, false, false);
    acc1 = __builtin_amdgcn_wmma_f32_16x16x32_f16(false, a, false, bf1, (short)0, acc1, false, false);
    acc2 = __builtin_amdgcn_wmma_f32_16x16x32_f16(false, a, false, bf2, (short)0, acc2, false, false);
    acc3 = __builtin_amdgcn_wmma_f32_16x16x32_f16(false, a, false, bf3, (short)0, acc3, false, false);
  }

  float bn0 = bias[nb +  0];
  float bn1 = bias[nb + 16];
  float bn2 = bias[nb + 32];
  float bn3 = bias[nb + 48];
#pragma unroll
  for (int r = 0; r < 8; ++r) {
    int mr = tile_m * 16 + r + half * 8;   // D layout: M = r + 8*(lane/16)
    float* orow = out + ((size_t)b * MO + mr) * COUT;
    orow[nb +  0] = acc0[r] + bn0;
    orow[nb + 16] = acc1[r] + bn1;
    orow[nb + 32] = acc2[r] + bn2;
    orow[nb + 48] = acc3[r] + bn3;
  }
}

// ---------------- 7: bilinear map -> token  (B,784,320)->(B,3136,320) ----------------
__global__ void map2token_kernel(const float* __restrict__ fmap,
                                 const float* __restrict__ pos,
                                 float* __restrict__ xt) {
  size_t t = (size_t)blockIdx.x * blockDim.x + threadIdx.x;
  size_t total = (size_t)B_ * N_ * (COUT / 4);
  if (t >= total) return;
  int q = (int)(t % (COUT / 4));
  size_t bn = t / (COUT / 4);
  int b = (int)(bn / N_);
  float gx = ((pos[bn * 2 + 0] + 1.f) * WO - 1.f) * 0.5f;
  float gy = ((pos[bn * 2 + 1] + 1.f) * HO - 1.f) * 0.5f;
  int x0 = (int)floorf(gx), y0 = (int)floorf(gy);
  float wx = gx - (float)x0, wy = gy - (float)y0;
  float4 a = {0.f, 0.f, 0.f, 0.f};
  const float* base = fmap + (size_t)b * MO * COUT + q * 4;
#pragma unroll
  for (int cy = 0; cy < 2; ++cy) {
#pragma unroll
    for (int cx = 0; cx < 2; ++cx) {
      int xi = x0 + cx, yi = y0 + cy;
      if (xi >= 0 && xi < WO && yi >= 0 && yi < HO) {
        float wgt = (cx ? wx : (1.f - wx)) * (cy ? wy : (1.f - wy));
        float4 v = *(const float4*)(base + (size_t)(yi * WO + xi) * COUT);
        a.x += wgt * v.x; a.y += wgt * v.y; a.z += wgt * v.z; a.w += wgt * v.w;
      }
    }
  }
  *(float4*)(xt + bn * COUT + q * 4) = a;
}

// ---------------- 8: LN + confidence + gumbel (ada tokens only) ----------------
__global__ __launch_bounds__(128) void ln_conf_kernel(
    const float* __restrict__ xt, const float* __restrict__ ln_g,
    const float* __restrict__ ln_b, const float* __restrict__ conf_w,
    const float* __restrict__ conf_b, const float* __restrict__ noise,
    float* __restrict__ pert) {
  int wid = (int)(((size_t)blockIdx.x * blockDim.x + threadIdx.x) >> 5);
  int lane = threadIdx.x & 31;
  if (wid >= B_ * NADA) return;
  int b = wid / NADA;
  int j = wid % NADA;
  const float* xp = xt + ((size_t)b * N_ + NGRID + j) * COUT;
  float s = 0.f, s2 = 0.f;
  for (int c = lane; c < COUT; c += 32) {
    float v = xp[c];
    s += v;
    s2 += v * v;
  }
#pragma unroll
  for (int o = 16; o > 0; o >>= 1) {
    s += __shfl_xor(s, o, 32);
    s2 += __shfl_xor(s2, o, 32);
  }
  float mu = s / (float)COUT;
  float var = s2 / (float)COUT - mu * mu;
  float rs = rsqrtf(var + 1e-5f);
  float cf = 0.f;
  for (int c = lane; c < COUT; c += 32) {
    float xn = (xp[c] - mu) * rs * ln_g[c] + ln_b[c];
    cf += xn * conf_w[c];
  }
#pragma unroll
  for (int o = 16; o > 0; o >>= 1) cf += __shfl_xor(cf, o, 32);
  if (lane == 0) {
    cf += conf_b[0];
    float nz = noise[(size_t)b * NADA + j];
    float g = -logf(-logf(nz + 1e-6f) + 1e-6f);
    pert[(size_t)b * NADA + j] = cf + g;  // T = 1
  }
}

// ---------------- 9: exact top-k via rank (stable, matches lax.top_k order) ----------
__global__ __launch_bounds__(256) void topk_kernel(const float* __restrict__ pert,
                                                   int* __restrict__ idxbuf) {
  __shared__ float sm[NADA];
  int b = blockIdx.x;
  const float* p = pert + (size_t)b * NADA;
  for (int i = threadIdx.x; i < NADA; i += 256) sm[i] = p[i];
  __syncthreads();
  for (int j = threadIdx.x; j < NADA; j += 256) {
    float vj = sm[j];
    int cnt = 0;
    for (int i = 0; i < NADA; ++i) {
      float vi = sm[i];
      cnt += (vi > vj) || (vi == vj && i < j);
    }
    if (cnt < SNUM) idxbuf[b * SNUM + cnt] = j;
  }
}

// ---------------- 10: gather outputs ----------------
__global__ void gather_x_kernel(const float* __restrict__ xt,
                                const int* __restrict__ idxbuf,
                                float* __restrict__ out_x) {
  size_t t = (size_t)blockIdx.x * blockDim.x + threadIdx.x;
  size_t total = (size_t)B_ * (NGRID + SNUM) * COUT;
  if (t >= total) return;
  int b = (int)(t / ((NGRID + SNUM) * COUT));
  int r = (int)(t % ((NGRID + SNUM) * COUT));
  int tok = r / COUT;
  int c = r % COUT;
  int sn = (tok < NGRID) ? tok : NGRID + idxbuf[b * SNUM + (tok - NGRID)];
  out_x[t] = xt[((size_t)b * N_ + sn) * COUT + c];
}

__global__ void gather_pos_kernel(const float* __restrict__ pos,
                                  const int* __restrict__ idxbuf,
                                  float* __restrict__ out_pos) {
  int t = blockIdx.x * blockDim.x + threadIdx.x;
  int total = B_ * (NGRID + SNUM) * 2;
  if (t >= total) return;
  int b = t / ((NGRID + SNUM) * 2);
  int r = t % ((NGRID + SNUM) * 2);
  int tok = r / 2;
  int d = r % 2;
  int sn = (tok < NGRID) ? tok : NGRID + idxbuf[b * SNUM + (tok - NGRID)];
  out_pos[t] = pos[((size_t)b * N_ + sn) * 2 + d];
}

// ---------------- host launcher ----------------
extern "C" void kernel_launch(void* const* d_in, const int* in_sizes, int n_in,
                              void* d_out, int out_size, void* d_ws, size_t ws_size,
                              hipStream_t stream) {
  (void)in_sizes; (void)n_in; (void)out_size; (void)ws_size;
  const float* x      = (const float*)d_in[0];
  const float* pos    = (const float*)d_in[1];
  const float* conv_w = (const float*)d_in[2];
  const float* conv_b = (const float*)d_in[3];
  const float* ln_g   = (const float*)d_in[4];
  const float* ln_b   = (const float*)d_in[5];
  const float* conf_w = (const float*)d_in[6];
  const float* conf_b = (const float*)d_in[7];
  const float* noise  = (const float*)d_in[8];

  char* ws = (char*)d_ws;
  size_t off = 0;
  auto take = [&](size_t bytes) -> void* {
    void* p = ws + off;
    off = (off + bytes + 255) & ~(size_t)255;
    return p;
  };
  const size_t sz_accum = (size_t)B_ * HW_ * C1;
  const size_t sz_feat  = (size_t)B_ * HW_ * CIN;
  const size_t sz_fmap2 = (size_t)B_ * MO * COUT;
  const size_t sz_xtok  = (size_t)B_ * N_ * COUT;

  float*    w_accum = (float*)take(sz_accum * 4);        // scatter accum, reused as h-filt out
  float*    w_nrm   = (float*)take(sz_accum * 4);        // normalized feat+mask
  _Float16* w_feat  = (_Float16*)take(sz_feat * 2);      // final NHWC feature map (f16)
  float*    w_fmap2 = (float*)take(sz_fmap2 * 4);        // conv output (B,784,320)
  float*    w_xtok  = (float*)take(sz_xtok * 4);         // sampled tokens (B,3136,320)
  float*    w_pert  = (float*)take((size_t)B_ * NADA * 4);
  int*      w_idx   = (int*)take((size_t)B_ * SNUM * 4);
  _Float16* w_pw    = (_Float16*)take((size_t)COUT * KC * 2);

  float* out_x   = (float*)d_out;
  float* out_pos = out_x + (size_t)B_ * (NGRID + SNUM) * COUT;

  // 0: zero the scatter accumulator
  zero_kernel<<<CDIV(sz_accum, 256), 256, 0, stream>>>(w_accum, sz_accum);
  // 1: scatter
  scatter_kernel<<<CDIV((size_t)B_ * N_ * C1, 256), 256, 0, stream>>>(x, pos, w_accum);
  // 2: normalize (accum -> nrm)
  normalize_kernel<<<CDIV(sz_accum, 256), 256, 0, stream>>>(w_accum, w_nrm);
  // 3: horizontal pass (nrm -> accum, reuse)
  hfilt_kernel<<<CDIV(sz_accum, 256), 256, 0, stream>>>(w_nrm, w_accum);
  // 4: vertical pass + combine -> f16 feat
  vfilt_combine_kernel<<<CDIV(sz_feat, 256), 256, 0, stream>>>(w_accum, w_nrm, w_feat);
  // 5: pack weights
  packw_kernel<<<CDIV(COUT * KC, 256), 256, 0, stream>>>(conv_w, w_pw);
  // 6: conv as WMMA GEMM, 16x64 tile per wave
  {
    dim3 grid((MO / 16) * (COUT / 64), B_);
    conv_wmma_kernel<<<grid, 32, 0, stream>>>(w_feat, w_pw, conv_b, w_fmap2);
  }
  // 7: bilinear sampling
  map2token_kernel<<<CDIV((size_t)B_ * N_ * (COUT / 4), 256), 256, 0, stream>>>(
      w_fmap2, pos, w_xtok);
  // 8: LN + conf + gumbel perturb
  {
    size_t waves = (size_t)B_ * NADA;
    ln_conf_kernel<<<CDIV(waves * 32, 128), 128, 0, stream>>>(
        w_xtok, ln_g, ln_b, conf_w, conf_b, noise, w_pert);
  }
  // 9: top-k by ranking
  topk_kernel<<<B_, 256, 0, stream>>>(w_pert, w_idx);
  // 10: gather outputs
  gather_x_kernel<<<CDIV((size_t)B_ * (NGRID + SNUM) * COUT, 256), 256, 0, stream>>>(
      w_xtok, w_idx, out_x);
  gather_pos_kernel<<<CDIV(B_ * (NGRID + SNUM) * 2, 256), 256, 0, stream>>>(
      pos, w_idx, out_pos);
}
